// Siamese_16939351015856
// MI455X (gfx1250) — matrix operand order repaired and checked by reading, not compiled, not run on hardware
//
#include <hip/hip_runtime.h>
#include <hip/hip_bf16.h>

// ---------------------------------------------------------------------------
// Siamese bi-GRU encoder for gfx1250 (MI455X), wave32 + WMMA bf16.
//   B=2048, L=128, D=64(vocab one-hot), H=128, gates G=3H=384.
// Pipeline:
//   1) cidx_kernel      : sequential scan -> compacted char index per (s,b,t)
//   2) cvt_linw         : lin_w f32 -> bf16 (for WMMA B operand)
//   3) gru_kernel       : 4 independent recurrences; w_hh(bf16)+w_ih(f32) in LDS,
//                         per-wave private 16-row hidden strip in LDS; 96 WMMAs/step
//   4) linear_kernel    : [2048,32768]x[32768,128] bf16 WMMA GEMM -> h0/h1 (f32)
//   5) dist_kernel      : exp(-sum|h0-h1|)
// ---------------------------------------------------------------------------

#define Bsz  2048
#define Lseq 128
#define Dvoc 64
#define Hd   128
#define G3   384

typedef __bf16 bf16_t;
typedef bf16_t v16bf __attribute__((ext_vector_type(16)));
typedef float  v8f   __attribute__((ext_vector_type(8)));

union FragBits { v16bf v; uint4 u[2]; };

// A fragment (16x32, 16-bit): lane r holds row r; 16 contiguous K
// starting at hi*16.  `row` points at [row][kchunk].
__device__ __forceinline__ v16bf load_a_frag(const bf16_t* row, int hi) {
  FragBits f;
  f.u[0] = *reinterpret_cast<const uint4*>(row + hi * 16);
  f.u[1] = *reinterpret_cast<const uint4*>(row + hi * 16 + 8);
  return f.v;
}

// B fragment (32x16, 16-bit): lane holds column n=lane&15; 16 contiguous K
// starting at hi*16.  `row` points at [n][kchunk] of the (N x K) weight.
__device__ __forceinline__ v16bf load_b_frag(const bf16_t* row, int hi) {
  FragBits f;
  const uint4* p = reinterpret_cast<const uint4*>(row + hi * 16);
  f.u[0] = p[0];
  f.u[1] = p[1];
  return f.v;
}

// ---------------------------------------------------------------------------
// Phase 0: compacted one-hot indices.  cidx[s][b][t] = char in [0,64) or -1.
// ---------------------------------------------------------------------------
__global__ void cidx_kernel(const int* __restrict__ seq0,
                            const int* __restrict__ seq1,
                            int* __restrict__ cidx) {
  int id = blockIdx.x * blockDim.x + threadIdx.x;
  if (id >= 2 * Bsz) return;
  int s = id >> 11, b = id & (Bsz - 1);
  const int* seq = (s ? seq1 : seq0) + (size_t)b * Lseq;
  int* out = cidx + ((size_t)(s * Bsz + b)) * Lseq;
  for (int t = 0; t < Lseq; ++t) out[t] = -1;
  int buf = 0, ended = 0;
  for (int j = 0; j < Lseq; ++j) {
    int ch = seq[j];
    if (ch == 30) buf++;          // inclusive cumsum of STARTs
    if (ch == 31) ended = 1;      // END itself is already masked
    int valid = (!ended) && (ch != 30);
    int pos = j - buf;
    pos = pos < 0 ? 0 : (pos > Lseq - 1 ? Lseq - 1 : pos);
    if (valid && ch >= 0 && ch < Dvoc) out[pos] = ch;
  }
}

// ---------------------------------------------------------------------------
// lin_w f32 -> bf16
// ---------------------------------------------------------------------------
__global__ void cvt_linw(const float* __restrict__ w, bf16_t* __restrict__ o) {
  int idx = blockIdx.x * blockDim.x + threadIdx.x;
  int stride = gridDim.x * blockDim.x;
  for (int i = idx; i < Hd * Lseq * 2 * Hd; i += stride)
    o[i] = (bf16_t)w[i];
}

// ---------------------------------------------------------------------------
// Phase 1: GRU recurrence.  grid=64: (s,dir) x 16 batch tiles of 128 rows.
// 8 waves/WG; each wave owns a private 16-row hidden strip -> barrier-free loop.
// ---------------------------------------------------------------------------
#define GRU_LDS_BYTES (98304 + 98304 + 1536 + 1536 + 32768 + 512)

__global__ __launch_bounds__(256)
void gru_kernel(const int* __restrict__ cidx,
                const float* __restrict__ w_ih_f, const float* __restrict__ w_hh_f,
                const float* __restrict__ b_ih_f, const float* __restrict__ b_hh_f,
                const float* __restrict__ w_ih_b, const float* __restrict__ w_hh_b,
                const float* __restrict__ b_ih_b, const float* __restrict__ b_hh_b,
                bf16_t* __restrict__ hcat) {
  extern __shared__ char smem[];
  bf16_t* whh  = (bf16_t*)smem;                                 // [384][128] bf16
  float*  wih  = (float*)(smem + 98304);                        // [384][64]  f32
  float*  bih  = (float*)(smem + 196608);                       // [384]
  float*  bhh  = (float*)(smem + 196608 + 1536);                // [384]
  bf16_t* hall = (bf16_t*)(smem + 196608 + 3072);               // 8 x [16][128]
  int*    call = (int*)(smem + 196608 + 3072 + 32768);          // 8 x [16]

  const int tid  = threadIdx.x;
  const int wave = tid >> 5, lane = tid & 31;
  const int r = lane & 15, hi = lane >> 4;
  const int problem = blockIdx.x >> 4;
  const int s = problem >> 1, dir = problem & 1;
  const int tile = blockIdx.x & 15;
  const int rowbase = tile * 128 + wave * 16;

  const float* Wih = dir ? w_ih_b : w_ih_f;
  const float* Whh = dir ? w_hh_b : w_hh_f;
  const float* Bih = dir ? b_ih_b : b_ih_f;
  const float* Bhh = dir ? b_hh_b : b_hh_f;

  for (int i = tid; i < G3 * Hd;   i += 256) whh[i] = (bf16_t)Whh[i];
  for (int i = tid; i < G3 * Dvoc; i += 256) wih[i] = Wih[i];
  for (int i = tid; i < G3;        i += 256) { bih[i] = Bih[i]; bhh[i] = Bhh[i]; }
  __syncthreads();

  bf16_t* hW = hall + wave * (16 * Hd);
  int*    cb = call + wave * 16;
  for (int i = lane; i < 16 * Hd; i += 32) hW[i] = (bf16_t)0.0f;

  const int* crow = cidx + ((size_t)(s * Bsz + rowbase)) * Lseq;
  bf16_t* dst0 = hcat + ((size_t)(s * Bsz + rowbase)) * (Lseq * 2 * Hd) + dir * Hd;

  for (int step = 0; step < Lseq; ++step) {
    const int t = dir ? (Lseq - 1 - step) : step;

    if (lane < 16) cb[lane] = crow[lane * Lseq + t];   // in-wave LDS: in-order
    int cm[8];
#pragma unroll
    for (int v = 0; v < 8; ++v) cm[v] = cb[v + hi * 8];

    v16bf afrag[4];
#pragma unroll
    for (int kk = 0; kk < 4; ++kk)
      afrag[kk] = load_a_frag(hW + r * Hd + kk * 32, hi);

    for (int nt = 0; nt < 8; ++nt) {
      v8f ar = {0, 0, 0, 0, 0, 0, 0, 0};
      v8f az = ar, an = ar;
#pragma unroll
      for (int kk = 0; kk < 4; ++kk) {
        v16bf br = load_b_frag(whh + (size_t)(nt * 16 + r) * Hd + kk * 32, hi);
        v16bf bz = load_b_frag(whh + (size_t)(Hd + nt * 16 + r) * Hd + kk * 32, hi);
        v16bf bn = load_b_frag(whh + (size_t)(2 * Hd + nt * 16 + r) * Hd + kk * 32, hi);
        ar = __builtin_amdgcn_wmma_f32_16x16x32_bf16(false, afrag[kk], false, br,
                                                     (short)0, ar, false, false);
        az = __builtin_amdgcn_wmma_f32_16x16x32_bf16(false, afrag[kk], false, bz,
                                                     (short)0, az, false, false);
        an = __builtin_amdgcn_wmma_f32_16x16x32_bf16(false, afrag[kk], false, bn,
                                                     (short)0, an, false, false);
      }
      const int col = nt * 16 + r;
#pragma unroll
      for (int v = 0; v < 8; ++v) {
        const int m = v + hi * 8;
        const int c = cm[v];
        float gr = ar[v] + bhh[col]          + bih[col];
        float gz = az[v] + bhh[Hd + col]     + bih[Hd + col];
        float hn = an[v] + bhh[2 * Hd + col];
        float xn = bih[2 * Hd + col];
        if (c >= 0) {
          gr += wih[col * Dvoc + c];
          gz += wih[(Hd + col) * Dvoc + c];
          xn += wih[(2 * Hd + col) * Dvoc + c];
        }
        float rg = 1.0f / (1.0f + expf(-gr));
        float zg = 1.0f / (1.0f + expf(-gz));
        float ng = tanhf(xn + rg * hn);
        float ho = (float)hW[m * Hd + col];
        float hv = (1.0f - zg) * ng + zg * ho;
        hW[m * Hd + col] = (bf16_t)hv;
      }
    }

    // cooperative strip write: 16 rows x 128 bf16 -> hcat[s][b][t*256 + dir*128 ..]
    bf16_t* dstp = dst0 + (size_t)t * (2 * Hd);
#pragma unroll
    for (int i = 0; i < 8; ++i) {
      int row = i * 2 + hi;
      *reinterpret_cast<uint4*>(dstp + (size_t)row * (Lseq * 2 * Hd) + r * 8) =
          *reinterpret_cast<const uint4*>(hW + row * Hd + r * 8);
    }
  }
}

// ---------------------------------------------------------------------------
// Phase 2: final linear, M=2048/seq, N=128, K=32768; LDS-tiled bf16 WMMA GEMM.
// grid=32: s x 16 M-tiles of 128 rows.  out_h = [h0 ; h1] f32 in d_out.
// ---------------------------------------------------------------------------
__global__ __launch_bounds__(256)
void linear_kernel(const bf16_t* __restrict__ hcat, const bf16_t* __restrict__ lw,
                   const float* __restrict__ lb, float* __restrict__ out_h) {
  __shared__ bf16_t Asm[128 * 64];
  __shared__ bf16_t Bsm[128 * 64];
  const int tid = threadIdx.x;
  const int wave = tid >> 5, lane = tid & 31, r = lane & 15, hi = lane >> 4;
  const int s = blockIdx.x >> 4, tile = blockIdx.x & 15;
  const int rowbase = tile * 128;
  const size_t K = (size_t)Lseq * 2 * Hd;  // 32768
  const bf16_t* Abase = hcat + ((size_t)(s * Bsz + rowbase)) * K;

  v8f acc[8];
#pragma unroll
  for (int nt = 0; nt < 8; ++nt) acc[nt] = (v8f){0, 0, 0, 0, 0, 0, 0, 0};

  for (int k0 = 0; k0 < (int)K; k0 += 64) {
#pragma unroll
    for (int it = 0; it < 4; ++it) {
      int u = it * 256 + tid;        // 1024 uint4 tiles of 8 bf16
      int row = u >> 3, k8 = u & 7;
      *reinterpret_cast<uint4*>(Asm + row * 64 + k8 * 8) =
          *reinterpret_cast<const uint4*>(Abase + (size_t)row * K + k0 + k8 * 8);
      *reinterpret_cast<uint4*>(Bsm + row * 64 + k8 * 8) =
          *reinterpret_cast<const uint4*>(lw + (size_t)row * K + k0 + k8 * 8);
    }
    __syncthreads();
#pragma unroll
    for (int kk = 0; kk < 2; ++kk) {
      v16bf a = load_a_frag(Asm + (wave * 16 + r) * 64 + kk * 32, hi);
#pragma unroll
      for (int nt = 0; nt < 8; ++nt) {
        v16bf bb = load_b_frag(Bsm + (nt * 16 + r) * 64 + kk * 32, hi);
        acc[nt] = __builtin_amdgcn_wmma_f32_16x16x32_bf16(false, a, false, bb,
                                                          (short)0, acc[nt], false, false);
      }
    }
    __syncthreads();
  }

  float* outp = out_h + (size_t)s * Bsz * Hd;
#pragma unroll
  for (int nt = 0; nt < 8; ++nt) {
    int col = nt * 16 + r;
    float bias = lb[col];
#pragma unroll
    for (int v = 0; v < 8; ++v) {
      int row = rowbase + wave * 16 + v + hi * 8;
      outp[(size_t)row * Hd + col] = acc[nt][v] + bias;
    }
  }
}

// ---------------------------------------------------------------------------
// Phase 3: distance
// ---------------------------------------------------------------------------
__global__ void dist_kernel(const float* __restrict__ h0, const float* __restrict__ h1,
                            float* __restrict__ dist) {
  int b = blockIdx.x * blockDim.x + threadIdx.x;
  if (b >= Bsz) return;
  float sum = 0.0f;
  for (int j = 0; j < Hd; ++j)
    sum += fabsf(h0[(size_t)b * Hd + j] - h1[(size_t)b * Hd + j]);
  dist[b] = expf(-sum);
}

// ---------------------------------------------------------------------------
extern "C" void kernel_launch(void* const* d_in, const int* in_sizes, int n_in,
                              void* d_out, int out_size, void* d_ws, size_t ws_size,
                              hipStream_t stream) {
  (void)in_sizes; (void)n_in; (void)out_size; (void)ws_size;
  const int*   seq0   = (const int*)d_in[0];
  const int*   seq1   = (const int*)d_in[1];
  const float* w_ih_f = (const float*)d_in[2];
  const float* w_hh_f = (const float*)d_in[3];
  const float* b_ih_f = (const float*)d_in[4];
  const float* b_hh_f = (const float*)d_in[5];
  const float* w_ih_b = (const float*)d_in[6];
  const float* w_hh_b = (const float*)d_in[7];
  const float* b_ih_b = (const float*)d_in[8];
  const float* b_hh_b = (const float*)d_in[9];
  const float* lin_w  = (const float*)d_in[10];
  const float* lin_b  = (const float*)d_in[11];

  char* ws = (char*)d_ws;
  int*    cidx  = (int*)ws;                                             // 2 MB
  bf16_t* hcat  = (bf16_t*)(ws + (size_t)(2 << 20));                    // 256 MB
  bf16_t* linwb = (bf16_t*)(ws + (size_t)(2 << 20) +
                            (size_t)2 * Bsz * Lseq * 2 * Hd * sizeof(bf16_t));

  float* out_dist = (float*)d_out;
  float* out_h    = out_dist + Bsz;           // h0 then h1, f32

  hipFuncSetAttribute((const void*)gru_kernel,
                      hipFuncAttributeMaxDynamicSharedMemorySize, GRU_LDS_BYTES);

  cidx_kernel<<<16, 256, 0, stream>>>(seq0, seq1, cidx);
  cvt_linw<<<1024, 256, 0, stream>>>(lin_w, linwb);
  gru_kernel<<<64, 256, GRU_LDS_BYTES, stream>>>(cidx,
      w_ih_f, w_hh_f, b_ih_f, b_hh_f,
      w_ih_b, w_hh_b, b_ih_b, b_hh_b, hcat);
  linear_kernel<<<32, 256, 0, stream>>>(hcat, linwb, lin_b, out_h);
  dist_kernel<<<16, 128, 0, stream>>>(out_h, out_h + (size_t)Bsz * Hd, out_dist);
}
